// GCN_30580167148117
// MI455X (gfx1250) — matrix operand order repaired and checked by reading, not compiled, not run on hardware
//
#include <hip/hip_runtime.h>
#include <hip/hip_bf16.h>
#include <math.h>

// ---------------------------------------------------------------------------
// GCN forward on MI455X (gfx1250, wave32).
// GEMMs run on v_wmma_f32_16x16x32_bf16 with 16x64 register blocking per wave
// (A fragment reused across 4 column tiles). Message passing is bf16-gather +
// f32 atomic scatter (bandwidth bound: ~3.7GB total -> ~160us @ 23.3TB/s).
// ---------------------------------------------------------------------------

typedef __attribute__((ext_vector_type(16))) __bf16 v16bf;
typedef __attribute__((ext_vector_type(8)))  __bf16 v8bf;
typedef __attribute__((ext_vector_type(8)))  float  v8f;

#define HDIM 256
#define GNUM 128
#define ONUM 10

// ------------------------------- small utils -------------------------------

__global__ void k_fill1(float* p, int n) {
    int i = blockIdx.x * blockDim.x + threadIdx.x;
    if (i < n) p[i] = 1.0f;
}

__global__ void k_deg_acc(const int* __restrict__ col, float* __restrict__ deg, int e) {
    int i = blockIdx.x * blockDim.x + threadIdx.x;
    if (i < e) atomicAdd(&deg[col[i]], 1.0f);
}

__global__ void k_rsqrt_inplace(float* d, int n) {
    int i = blockIdx.x * blockDim.x + threadIdx.x;
    if (i < n) d[i] = rsqrtf(d[i]);
}

__global__ void k_f32_to_bf16(const float* __restrict__ src, __bf16* __restrict__ dst, int n) {
    int i = blockIdx.x * blockDim.x + threadIdx.x;
    if (i < n) dst[i] = (__bf16)src[i];
}

// Wt[j*K + k] = bf16(W[k*HOUT + j]) : transpose so B-fragment loads are
// contiguous along K (32B per lane per k-step).
__global__ void k_transposeW(const float* __restrict__ W, __bf16* __restrict__ Wt,
                             int K, int Hout) {
    int idx = blockIdx.x * blockDim.x + threadIdx.x;
    if (idx < K * Hout) {
        int k = idx / Hout, j = idx - k * Hout;
        Wt[(size_t)j * K + k] = (__bf16)W[idx];
    }
}

// ------------------------------- WMMA GEMM ---------------------------------
// C[M x HDIM] (bf16) = A[M x K] (bf16, row major) * Wt^T  (Wt is [HDIM x K]).
// 256 threads = 8 waves; each wave computes a 16x64 strip = 4 WMMA tiles,
// reusing the A fragment across the 4 column tiles each k-step.
// grid = (HDIM/64, ceil(M/16/8)).
// A-fragment (16x32 bf16): lane<16 -> row=lane, K = [0..15];
// lane>=16 -> row=lane-16, K = [16..31] (relative to k-step base).
// B-fragment (32x16): lane<16 -> col=lane, K=[0..15]; lane>=16 -> K=[16..31].
// C/D: VGPR r -> row r (lanes 0-15) / r+8 (lanes 16-31), col = lane&15.
template <int K>
__global__ __launch_bounds__(256) void k_gemm_bf16(const __bf16* __restrict__ A,
                                                   const __bf16* __restrict__ Wt,
                                                   __bf16* __restrict__ C, int M) {
    const int lane    = threadIdx.x & 31;
    const int wave    = threadIdx.x >> 5;
    const int rowTile = blockIdx.y * 8 + wave;
    if (rowTile * 16 + 16 > M) return;   // wave-uniform exit: EXEC stays all-ones
    const int row0    = rowTile * 16;
    const int colBase = blockIdx.x * 64;

    const int aRow = row0 + (lane & 15);
    const int aK   = (lane >> 4) * 16;     // 0 or 16
    const int bCol = colBase + (lane & 15);
    const int bK   = (lane >> 4) * 16;     // 0 or 16

    const __bf16* aPtr = A  + (size_t)aRow * K + aK;
    const __bf16* bPtr = Wt + (size_t)bCol * K + bK;

    v8f c0 = {}, c1 = {}, c2 = {}, c3 = {};
#pragma unroll
    for (int ks = 0; ks < K; ks += 32) {
        // A fragment: loaded once, reused by 4 WMMAs
        v16bf a = *(const v16bf*)(aPtr + ks);       // K = base+[0..15] (or +16)
        // 4 B fragments: column tiles colBase + {0,16,32,48}
        v16bf b0 = *(const v16bf*)(bPtr + ks);
        v16bf b1 = *(const v16bf*)(bPtr + (size_t)16 * K + ks);
        v16bf b2 = *(const v16bf*)(bPtr + (size_t)32 * K + ks);
        v16bf b3 = *(const v16bf*)(bPtr + (size_t)48 * K + ks);
        c0 = __builtin_amdgcn_wmma_f32_16x16x32_bf16(false, a, false, b0, (short)0, c0, false, false);
        c1 = __builtin_amdgcn_wmma_f32_16x16x32_bf16(false, a, false, b1, (short)0, c1, false, false);
        c2 = __builtin_amdgcn_wmma_f32_16x16x32_bf16(false, a, false, b2, (short)0, c2, false, false);
        c3 = __builtin_amdgcn_wmma_f32_16x16x32_bf16(false, a, false, b3, (short)0, c3, false, false);
    }

    const int mBase = row0 + ((lane >> 4) << 3);    // +0 or +8
    __bf16* out = C + (size_t)mBase * HDIM + bCol;
#pragma unroll
    for (int r = 0; r < 8; ++r) {
        out[(size_t)r * HDIM +  0] = (__bf16)c0[r];
        out[(size_t)r * HDIM + 16] = (__bf16)c1[r];
        out[(size_t)r * HDIM + 32] = (__bf16)c2[r];
        out[(size_t)r * HDIM + 48] = (__bf16)c3[r];
    }
}

// --------------------------- message passing -------------------------------

// agg[i,f] = dinv[i]^2 * hw[i,f]   (self-loop term; also zero-initializes agg)
__global__ void k_init_agg(const __bf16* __restrict__ hw, const float* __restrict__ dinv,
                           float* __restrict__ agg, int total) {
    int i = blockIdx.x * blockDim.x + threadIdx.x;
    if (i < total) {
        float d = dinv[i >> 8];
        agg[i] = d * d * (float)hw[i];
    }
}

// One wave per edge, 8 features per lane: agg[col] += norm * hw[row]
__global__ __launch_bounds__(256) void k_scatter(const int* __restrict__ row,
                                                 const int* __restrict__ col,
                                                 const float* __restrict__ dinv,
                                                 const __bf16* __restrict__ hw,
                                                 float* __restrict__ agg, int e) {
    int edge = blockIdx.x * 8 + (threadIdx.x >> 5);
    if (edge >= e) return;
    int lane = threadIdx.x & 31;
    int r = row[edge], c = col[edge];
    float norm = dinv[r] * dinv[c];
    v8bf v = *(const v8bf*)(hw + (size_t)r * HDIM + lane * 8);
    float* dst = agg + (size_t)c * HDIM + lane * 8;
#pragma unroll
    for (int i = 0; i < 8; ++i) atomicAdd(dst + i, (float)v[i] * norm);
}

__global__ void k_relu_to_bf16(const float* __restrict__ agg, __bf16* __restrict__ h, int total) {
    int i = blockIdx.x * blockDim.x + threadIdx.x;
    if (i < total) h[i] = (__bf16)fmaxf(agg[i], 0.0f);
}

// ------------------------------- pooling -----------------------------------

__global__ void k_count(const int* __restrict__ batch, float* __restrict__ counts, int n) {
    int i = blockIdx.x * blockDim.x + threadIdx.x;
    if (i < n) atomicAdd(&counts[batch[i]], 1.0f);
}

__global__ void k_poolsum(const __bf16* __restrict__ h, const int* __restrict__ batch,
                          float* __restrict__ pooled, int total) {
    int i = blockIdx.x * blockDim.x + threadIdx.x;
    if (i < total) {
        int node = i >> 8, f = i & 255;
        atomicAdd(&pooled[batch[node] * HDIM + f], (float)h[i]);
    }
}

__global__ void k_poolnorm(float* __restrict__ pooled, const float* __restrict__ counts, int total) {
    int i = blockIdx.x * blockDim.x + threadIdx.x;
    if (i < total) pooled[i] /= fmaxf(counts[i >> 8], 1.0f);
}

// ------------------------------- MLP head ----------------------------------

__global__ void k_mlp1(const float* __restrict__ pooled, const float* __restrict__ Wm1,
                       const float* __restrict__ bm1, float* __restrict__ hmid) {
    int i = blockIdx.x * blockDim.x + threadIdx.x;   // i < G*H
    int g = i >> 8, j = i & 255;
    const float* p = pooled + (size_t)g * HDIM;
    float acc = bm1[j];
    for (int k = 0; k < HDIM; ++k) acc = fmaf(p[k], Wm1[(size_t)k * HDIM + j], acc);
    hmid[i] = fmaxf(acc, 0.0f);
}

__global__ void k_mlp2(const float* __restrict__ hmid, const float* __restrict__ Wm2,
                       const float* __restrict__ bm2, float* __restrict__ last, int total) {
    int i = blockIdx.x * blockDim.x + threadIdx.x;   // i < G*OUT
    if (i < total) {
        int g = i / ONUM, o = i - g * ONUM;
        const float* hp = hmid + (size_t)g * HDIM;
        float acc = bm2[o];
        for (int k = 0; k < HDIM; ++k) acc = fmaf(hp[k], Wm2[(size_t)k * ONUM + o], acc);
        last[i] = acc;
    }
}

// out = [log_softmax(last) | sigmoid(last) | last], each G*OUT
__global__ void k_head(const float* __restrict__ last, float* __restrict__ out) {
    int g = threadIdx.x;
    if (g < GNUM) {
        const float* v = last + g * ONUM;
        float m = v[0];
        for (int o = 1; o < ONUM; ++o) m = fmaxf(m, v[o]);
        float s = 0.0f;
        for (int o = 0; o < ONUM; ++o) s += expf(v[o] - m);
        float lse = m + logf(s);
        for (int o = 0; o < ONUM; ++o) {
            float x = v[o];
            out[g * ONUM + o]                   = x - lse;
            out[GNUM * ONUM + g * ONUM + o]     = 1.0f / (1.0f + expf(-x));
            out[2 * GNUM * ONUM + g * ONUM + o] = x;
        }
    }
}

// ------------------------------- launcher ----------------------------------

static inline size_t align256(size_t x) { return (x + 255) & ~(size_t)255; }

extern "C" void kernel_launch(void* const* d_in, const int* in_sizes, int n_in,
                              void* d_out, int out_size, void* d_ws, size_t ws_size,
                              hipStream_t stream) {
    const int IN = 128;
    const int N  = in_sizes[0] / IN;
    const int E  = in_sizes[1] / 2;

    const float* x    = (const float*)d_in[0];
    const int*   erow = (const int*)d_in[1];          // edge_index[0]
    const int*   ecol = ((const int*)d_in[1]) + E;    // edge_index[1]
    const int*   batch = (const int*)d_in[3];
    const float* W[3] = { (const float*)d_in[4], (const float*)d_in[5], (const float*)d_in[6] };
    const float* Wm1 = (const float*)d_in[7];
    const float* bm1 = (const float*)d_in[8];
    const float* Wm2 = (const float*)d_in[9];
    const float* bm2 = (const float*)d_in[10];
    float* out = (float*)d_out;

    // ---- workspace carve-up (all 256B aligned), ~103 MB total ----
    char* ws = (char*)d_ws;
    size_t off = 0;
    float*  dinv  = (float*)(ws + off);  off += align256((size_t)N * 4);
    __bf16* Wt    = (__bf16*)(ws + off); off += align256((size_t)HDIM * HDIM * 2);
    __bf16* hbf   = (__bf16*)(ws + off); off += align256((size_t)N * HDIM * 2);
    __bf16* hwbf  = (__bf16*)(ws + off); off += align256((size_t)N * HDIM * 2);
    float*  agg   = (float*)(ws + off);  off += align256((size_t)N * HDIM * 4);
    float*  pooled= (float*)(ws + off);  off += align256((size_t)GNUM * HDIM * 4);
    float*  counts= (float*)(ws + off);  off += align256((size_t)GNUM * 4);
    float*  hmid  = (float*)(ws + off);  off += align256((size_t)GNUM * HDIM * 4);
    float*  last  = (float*)(ws + off);  off += align256((size_t)GNUM * ONUM * 4);
    (void)ws_size; (void)n_in; (void)out_size;

    const int T = 256;
    // zero pool accumulators
    hipMemsetAsync(pooled, 0, ((size_t)GNUM * HDIM) * 4, stream);
    hipMemsetAsync(counts, 0, (size_t)GNUM * 4, stream);

    // degree (self-loop included) -> dinv_sqrt
    k_fill1<<<(N + T - 1) / T, T, 0, stream>>>(dinv, N);
    k_deg_acc<<<(E + T - 1) / T, T, 0, stream>>>(ecol, dinv, E);
    k_rsqrt_inplace<<<(N + T - 1) / T, T, 0, stream>>>(dinv, N);

    // x -> bf16 (layer-0 GEMM input)
    k_f32_to_bf16<<<((size_t)N * IN + T - 1) / T, T, 0, stream>>>(x, hbf, N * IN);

    const int rowBlocks = ((N + 15) / 16 + 7) / 8;
    const dim3 ggrid(HDIM / 64, rowBlocks);

    for (int l = 0; l < 3; ++l) {
        const int K = (l == 0) ? IN : HDIM;
        k_transposeW<<<((size_t)K * HDIM + T - 1) / T, T, 0, stream>>>(W[l], Wt, K, HDIM);
        if (K == IN)
            k_gemm_bf16<128><<<ggrid, T, 0, stream>>>(hbf, Wt, hwbf, N);
        else
            k_gemm_bf16<256><<<ggrid, T, 0, stream>>>(hbf, Wt, hwbf, N);
        k_init_agg<<<((size_t)N * HDIM + T - 1) / T, T, 0, stream>>>(hwbf, dinv, agg, N * HDIM);
        k_scatter<<<(E + 7) / 8, T, 0, stream>>>(erow, ecol, dinv, hwbf, agg, E);
        k_relu_to_bf16<<<((size_t)N * HDIM + T - 1) / T, T, 0, stream>>>(agg, hbf, N * HDIM);
    }

    // global mean pool
    k_count<<<(N + T - 1) / T, T, 0, stream>>>(batch, counts, N);
    k_poolsum<<<((size_t)N * HDIM + T - 1) / T, T, 0, stream>>>(hbf, batch, pooled, N * HDIM);
    k_poolnorm<<<(GNUM * HDIM + T - 1) / T, T, 0, stream>>>(pooled, counts, GNUM * HDIM);

    // MLP head + output transforms
    k_mlp1<<<(GNUM * HDIM) / T, T, 0, stream>>>(pooled, Wm1, bm1, hmid);
    k_mlp2<<<(GNUM * ONUM + T - 1) / T, T, 0, stream>>>(hmid, Wm2, bm2, last, GNUM * ONUM);
    k_head<<<1, GNUM, 0, stream>>>(last, out);
}
